// GeneAttentionLayer_5531917877943
// MI455X (gfx1250) — compile-verified, matrix operand order checked by reading, not run on hardware
//
#include <hip/hip_runtime.h>
#include <hip/hip_bf16.h>

// ---------------------------------------------------------------------------
// GeneAttentionLayer for MI455X (gfx1250, wave32, WMMA).
// Pipeline:
//   1) cast x, W_qk (rows 0..2D of in_proj_w), out_w -> bf16; V -> Vt[B][H][HD][S] bf16
//   2) QK = x @ Wqk^T (+in_proj_b)  : bf16 WMMA GEMM, out bf16 [4096 x 2048]
//   3) flash masked-renorm attention: per-wave 16-query tile, online
//      out = sum(e^{s-m} M V) / (sum(e^{s-m} M) + EPS * sum(e^{s-m}))
//      row-sums Zm / Z computed by WMMA against a ones operand (replicated
//      C-layout), only the row-max uses lane shuffles.
//   4) out = attn @ out_w^T + out_b : bf16 WMMA GEMM, out fp32
// ---------------------------------------------------------------------------

typedef __bf16 bf16;
typedef __attribute__((ext_vector_type(16))) __bf16 v16bf;
typedef __attribute__((ext_vector_type(8)))  __bf16 v8bf;
typedef __attribute__((ext_vector_type(8)))  float  v8f;

constexpr int kB  = 2;
constexpr int kS  = 2048;
constexpr int kD  = 1024;
constexpr int kH  = 16;
constexpr int kHD = 64;
constexpr int kBS = kB * kS;            // 4096 total rows
constexpr float kEPS   = 1e-8f;
constexpr float kScale = 0.125f;        // 1/sqrt(64)

// Load one 16x32 (bf16) A/B WMMA fragment for this lane.
// Lane layout (wave32): row/col = lane&15; lanes 0-15 cover K {0..7,16..23},
// lanes 16-31 cover K {8..15,24..31}. Two contiguous 16B chunks per lane.
__device__ __forceinline__ v16bf ld_frag(const bf16* p, int kh) {
  v8bf lo = *reinterpret_cast<const v8bf*>(p + kh);
  v8bf hi = *reinterpret_cast<const v8bf*>(p + 16 + kh);
  v16bf f;
#pragma unroll
  for (int i = 0; i < 8; ++i) { f[i] = lo[i]; f[i + 8] = hi[i]; }
  return f;
}

__device__ __forceinline__ v8f wmma_bf16(v16bf a, v16bf b, v8f c) {
  return __builtin_amdgcn_wmma_f32_16x16x32_bf16(false, a, false, b,
                                                 (short)0, c, false, false);
}

// ---------------------------------------------------------------------------
// casts / transpose
// ---------------------------------------------------------------------------
__global__ void cast_f32_bf16(const float* __restrict__ in,
                              bf16* __restrict__ out, int n) {
  int i = blockIdx.x * blockDim.x + threadIdx.x;
  if (i < n) out[i] = (bf16)in[i];
}

// V[b][t][h][d] (f32) -> Vt[b][h][d][t] (bf16): PV B-operand becomes
// K-contiguous per lane.
__global__ void transpose_v_bf16(const float* __restrict__ V,
                                 bf16* __restrict__ Vt) {
  int i = blockIdx.x * blockDim.x + threadIdx.x;   // < B*S*H*HD
  int d = i & (kHD - 1);
  int h = (i >> 6) & (kH - 1);
  int t = (i >> 10) & (kS - 1);
  int b = i >> 21;
  Vt[(((size_t)(b * kH + h)) * kHD + d) * kS + t] = (bf16)V[i];
}

// ---------------------------------------------------------------------------
// NT GEMM: C[M x N] = A[M x K] * B[N x K]^T + bias, bf16 inputs, f32 accum.
// 128x128 block tile, 8 waves of 64x32 (4x2 accumulators of 16x16).
// Operands are L2-resident (192 MB L2 >> working set): fragments load
// directly from global (two b128 per fragment), prefetch hints next K-slab.
// ---------------------------------------------------------------------------
template <bool OUT_BF16>
__global__ __launch_bounds__(256)
void gemm_nt_bf16(const bf16* __restrict__ A, const bf16* __restrict__ Bw,
                  const float* __restrict__ bias, void* __restrict__ Cout,
                  int Nn, int Kk) {
  const int lane = threadIdx.x & 31;
  const int wid  = threadIdx.x >> 5;
  const int wm = wid >> 2;               // 0..1 : 64-row group
  const int wn = wid & 3;                // 0..3 : 32-col group
  const int n16 = lane & 15;
  const int hi  = (lane >> 4) & 1;
  const int kh  = hi ? 8 : 0;

  const int m0 = blockIdx.y * 128 + wm * 64;
  const int n0 = blockIdx.x * 128 + wn * 32;

  v8f acc[4][2];
#pragma unroll
  for (int i = 0; i < 4; ++i)
#pragma unroll
    for (int j = 0; j < 2; ++j) acc[i][j] = {};

  const bf16* aptr[4];
  const bf16* bptr[2];
#pragma unroll
  for (int i = 0; i < 4; ++i) aptr[i] = A + (size_t)(m0 + i * 16 + n16) * Kk;
#pragma unroll
  for (int j = 0; j < 2; ++j) bptr[j] = Bw + (size_t)(n0 + j * 16 + n16) * Kk;

  for (int k = 0; k < Kk; k += 32) {
    v16bf af[4], bfr[2];
#pragma unroll
    for (int i = 0; i < 4; ++i) af[i] = ld_frag(aptr[i] + k, kh);
#pragma unroll
    for (int j = 0; j < 2; ++j) bfr[j] = ld_frag(bptr[j] + k, kh);
    if (k + 128 < Kk) {                      // hint next K-slab (L2 resident)
      __builtin_prefetch(aptr[0] + k + 128, 0, 1);
      __builtin_prefetch(bptr[0] + k + 128, 0, 1);
    }
#pragma unroll
    for (int i = 0; i < 4; ++i)
#pragma unroll
      for (int j = 0; j < 2; ++j)
        acc[i][j] = wmma_bf16(af[i], bfr[j], acc[i][j]);
  }

#pragma unroll
  for (int i = 0; i < 4; ++i) {
    const int mrow0 = m0 + i * 16 + hi * 8;   // C layout: vgpr r -> row r(+8)
#pragma unroll
    for (int j = 0; j < 2; ++j) {
      const int col = n0 + j * 16 + n16;      // C layout: lane&15 -> col
      const float bv = bias ? bias[col] : 0.0f;
#pragma unroll
      for (int r = 0; r < 8; ++r) {
        float v = acc[i][j][r] + bv;
        size_t off = (size_t)(mrow0 + r) * Nn + col;
        if (OUT_BF16) ((bf16*)Cout)[off] = (bf16)v;
        else          ((float*)Cout)[off] = v;
      }
    }
  }
}

// ---------------------------------------------------------------------------
// Flash masked-renormalized attention. One wave per (b, h, 16-query tile).
// QK: [4096 x 2048] bf16, Q in cols [0,1024), K in cols [1024,2048).
// Vt: [B][H][HD][S] bf16.  M: [B][S][S] f32.  Oattn: [4096 x 1024] bf16.
// Per 32-key chunk: 4 WMMA (Q.K^T), 4 WMMA (P.V), 2 WMMA (row sums Zm / Z
// against ones operand); lane shuffles only for the running row-max.
// ---------------------------------------------------------------------------
__global__ __launch_bounds__(256)
void flash_mask_attn(const bf16* __restrict__ QK, const bf16* __restrict__ Vt,
                     const float* __restrict__ Mmask,
                     bf16* __restrict__ Oattn) {
  // per-wave 16x64 tile: cols [0,32) masked P, cols [32,64) unmasked P
  __shared__ alignas(16) bf16 ldsP[8][16 * 64];

  const int lane = threadIdx.x & 31;
  const int wid  = threadIdx.x >> 5;
  const int n16 = lane & 15;
  const int hi  = (lane >> 4) & 1;
  const int kh  = hi ? 8 : 0;

  const int gw   = blockIdx.x * 8 + wid;          // 0..4095
  const int tile = gw & 127;                      // S/16 tiles per (b,h)
  const int bh   = gw >> 7;
  const int h    = bh & (kH - 1);
  const int b    = bh >> 4;
  const int q0   = tile * 16;

  // ones operand for row-sum WMMAs
  v16bf ones;
#pragma unroll
  for (int i = 0; i < 16; ++i) ones[i] = (bf16)1.0f;

  // Q fragments: row = q0 + (lane&15), 64 contiguous bf16 (head h)
  const bf16* qrow = QK + (size_t)(b * kS + q0 + n16) * (2 * kD) + h * kHD;
  const v16bf qa0 = ld_frag(qrow, kh);
  const v16bf qa1 = ld_frag(qrow + 32, kh);

  float row_m[8], z_all[8], z_m[8];
  v8f acc[4];
#pragma unroll
  for (int r = 0; r < 8; ++r) { row_m[r] = -1e30f; z_all[r] = 0.f; z_m[r] = 0.f; }
#pragma unroll
  for (int nt = 0; nt < 4; ++nt) acc[nt] = {};

  const float* Mbase = Mmask + (size_t)b * kS * kS + (size_t)q0 * kS;
  const bf16*  kbase = QK + kD + h * kHD;                       // K half
  const bf16*  vbase = Vt + (size_t)(b * kH + h) * kHD * kS;

  for (int kc = 0; kc < kS; kc += 32) {
    // ---- scores: two 16x16 tiles, K-dim = HD = 64 (2 wmma steps each) ----
    const bf16* k0p = kbase + (size_t)(b * kS + kc + n16) * (2 * kD);
    const bf16* k1p = k0p + (size_t)16 * (2 * kD);
    v8f s0 = {}, s1 = {};
    s0 = wmma_bf16(qa0, ld_frag(k0p, kh),      s0);
    s0 = wmma_bf16(qa1, ld_frag(k0p + 32, kh), s0);
    s1 = wmma_bf16(qa0, ld_frag(k1p, kh),      s1);
    s1 = wmma_bf16(qa1, ld_frag(k1p + 32, kh), s1);

    // ---- online max + exp; stage masked & unmasked P in LDS ----
    float corr[8];
#pragma unroll
    for (int r = 0; r < 8; ++r) {
      const int mrow = r + hi * 8;                 // row within 16-query tile
      float sc0 = s0[r] * kScale;
      float sc1 = s1[r] * kScale;
      float mk0 = Mbase[(size_t)mrow * kS + kc + n16];
      float mk1 = Mbase[(size_t)mrow * kS + kc + 16 + n16];
      float mx = fmaxf(sc0, sc1);
      mx = fmaxf(mx, __shfl_xor(mx, 1));
      mx = fmaxf(mx, __shfl_xor(mx, 2));
      mx = fmaxf(mx, __shfl_xor(mx, 4));
      mx = fmaxf(mx, __shfl_xor(mx, 8));
      float mnew = fmaxf(row_m[r], mx);
      float c = __expf(row_m[r] - mnew);
      row_m[r] = mnew;
      corr[r] = c;
      float p0 = __expf(sc0 - mnew);
      float p1 = __expf(sc1 - mnew);
      float pm0 = p0 * mk0, pm1 = p1 * mk1;
      bf16* prow = &ldsP[wid][mrow * 64];
      prow[n16]      = (bf16)pm0;     // masked P, cols [0,32)
      prow[16 + n16] = (bf16)pm1;
      prow[32 + n16] = (bf16)p0;      // unmasked P, cols [32,64)
      prow[48 + n16] = (bf16)p1;
    }

    // reload P tiles as A-fragments (C-layout -> A-layout via LDS)
    const bf16* pbase = &ldsP[wid][(size_t)n16 * 64];
    const v16bf pfm = ld_frag(pbase, kh);        // masked
    const v16bf pfu = ld_frag(pbase + 32, kh);   // unmasked

    // row sums via matrix core: every lane's element r = row sum (replicated)
    v8f zm_t = {}, za_t = {};
    zm_t = wmma_bf16(pfm, ones, zm_t);
    za_t = wmma_bf16(pfu, ones, za_t);

#pragma unroll
    for (int r = 0; r < 8; ++r) {
      z_m[r]   = z_m[r]   * corr[r] + zm_t[r];
      z_all[r] = z_all[r] * corr[r] + za_t[r];
    }

    // rescale existing accumulator by exp(m_old - m_new)
#pragma unroll
    for (int nt = 0; nt < 4; ++nt)
#pragma unroll
      for (int r = 0; r < 8; ++r) acc[nt][r] *= corr[r];

    // ---- P (16x32) x V (32x64) ----
#pragma unroll
    for (int nt = 0; nt < 4; ++nt) {
      v16bf vf = ld_frag(vbase + (size_t)(nt * 16 + n16) * kS + kc, kh);
      acc[nt] = wmma_bf16(pfm, vf, acc[nt]);
    }
  }

  // out = acc / (Zm + EPS * Z)   (Z-shift by running max cancels exactly)
#pragma unroll
  for (int r = 0; r < 8; ++r) {
    const float inv = 1.0f / (z_m[r] + kEPS * z_all[r]);
    const int mrow = r + hi * 8;
    const size_t rowoff = (size_t)(b * kS + q0 + mrow) * kD + h * kHD;
#pragma unroll
    for (int nt = 0; nt < 4; ++nt)
      Oattn[rowoff + nt * 16 + n16] = (bf16)(acc[nt][r] * inv);
  }
}

// ---------------------------------------------------------------------------
extern "C" void kernel_launch(void* const* d_in, const int* in_sizes, int n_in,
                              void* d_out, int out_size, void* d_ws, size_t ws_size,
                              hipStream_t stream) {
  (void)in_sizes; (void)n_in; (void)out_size; (void)ws_size;
  const float* x     = (const float*)d_in[0];   // [B,S,D]
  const float* V     = (const float*)d_in[1];   // [B,S,H,HD]
  const float* M     = (const float*)d_in[2];   // [B,S,S]
  const float* in_w  = (const float*)d_in[3];   // [3D,D]
  const float* in_b  = (const float*)d_in[4];   // [3D]
  const float* out_w = (const float*)d_in[5];   // [D,D]
  const float* out_b = (const float*)d_in[6];   // [D]
  float* out = (float*)d_out;

  char* ws = (char*)d_ws;
  bf16* xb   = (bf16*)ws;  ws += (size_t)kBS * kD * 2;        // 8 MiB
  bf16* wqk  = (bf16*)ws;  ws += (size_t)2 * kD * kD * 2;     // 4 MiB
  bf16* wob  = (bf16*)ws;  ws += (size_t)kD * kD * 2;         // 2 MiB
  bf16* qk   = (bf16*)ws;  ws += (size_t)kBS * 2 * kD * 2;    // 16 MiB
  bf16* vt   = (bf16*)ws;  ws += (size_t)kB * kD * kS * 2;    // 8 MiB
  bf16* attn = (bf16*)ws;                                     // 8 MiB

  const int nX = kBS * kD;            // 4 Mi
  const int nWqk = 2 * kD * kD;       // 2 Mi (rows 0..2D of in_proj_w)
  const int nWo = kD * kD;            // 1 Mi

  cast_f32_bf16<<<(nX + 255) / 256, 256, 0, stream>>>(x, xb, nX);
  cast_f32_bf16<<<(nWqk + 255) / 256, 256, 0, stream>>>(in_w, wqk, nWqk);
  cast_f32_bf16<<<(nWo + 255) / 256, 256, 0, stream>>>(out_w, wob, nWo);
  transpose_v_bf16<<<nX / 256, 256, 0, stream>>>(V, vt);

  // QK = x @ Wqk^T + b_qk  -> [4096 x 2048] bf16
  gemm_nt_bf16<true><<<dim3(2 * kD / 128, kBS / 128), 256, 0, stream>>>(
      xb, wqk, in_b, qk, 2 * kD, kD);

  // flash masked-renorm attention -> attn [4096 x 1024] bf16
  flash_mask_attn<<<kBS * kH / (16 * 8), 256, 0, stream>>>(qk, vt, M, attn);

  // out = attn @ out_w^T + out_b -> fp32
  gemm_nt_bf16<false><<<dim3(kD / 128, kBS / 128), 256, 0, stream>>>(
      attn, wob, out_b, out, kD, kD);
}
